// GodAC_31911607009638
// MI455X (gfx1250) — matrix operand order repaired and checked by reading, not compiled, not run on hardware
//
#include <hip/hip_runtime.h>
#include <hip/hip_bf16.h>
#include <math.h>

// ---------------------------------------------------------------------------
// Sizes (compile-time constants from the reference)
// ---------------------------------------------------------------------------
#define NAG   512           // agents
#define DHID  256           // hidden
#define DOBS  128           // obs
#define NPAIR 130816        // 512*511/2

typedef __attribute__((ext_vector_type(16))) _Float16 v16h;
typedef __attribute__((ext_vector_type(8)))  _Float16 v8h;
typedef __attribute__((ext_vector_type(8)))  float    v8f;

// ---------------------------------------------------------------------------
// Fragment loaders (wave32, v_wmma_f32_16x16x32_f16 operand layouts)
// A 16x32 (MxK): lane m=l%16, half=l/16; K chunks at half*8 and 16+half*8
// B 32x16 (KxN): lane n=l%16, half=l/16; K chunks at half*8 and 16+half*8
// Caller pre-offsets the row pointer by the half-dependent amount.
// ---------------------------------------------------------------------------
__device__ inline v16h ld_fragA(const _Float16* __restrict__ p) {
    v8h lo = *(const v8h*)p;          // K = half*8 + 0..7
    v8h hi = *(const v8h*)(p + 16);   // K = 16 + half*8 + 0..7
    v16h f;
#pragma unroll
    for (int i = 0; i < 8; ++i) { f[i] = lo[i]; f[i + 8] = hi[i]; }
    return f;
}

__device__ inline v16h ld_fragB(const _Float16* __restrict__ p) {
    v8h lo = *(const v8h*)p;          // K = half*8 + 0..7
    v8h hi = *(const v8h*)(p + 16);   // K = 16 + half*8 + 0..7
    v16h f;
#pragma unroll
    for (int i = 0; i < 8; ++i) { f[i] = lo[i]; f[i + 8] = hi[i]; }
    return f;
}

// ---------------------------------------------------------------------------
// Generic wave-tile GEMM: Y[m,n] = act(scale * sum_k X[m,k]*W[n,k] + bias[n])
// One 32-thread wave (= one block) per 16x16 output tile.
// grid = (N/16, M/16). K must be a multiple of 32. Rows 16B-aligned.
// ---------------------------------------------------------------------------
__global__ __launch_bounds__(32)
void gac_wmma_gemm_xwT(const _Float16* __restrict__ X, int ldx,
                       const _Float16* __restrict__ W, int ldw,
                       const float* __restrict__ bias,
                       float* __restrict__ Yf, _Float16* __restrict__ Yh, int ldy,
                       int K, float scale, int relu)
{
    const int lane  = threadIdx.x & 31;
    const int lo    = lane & 15;
    const int half  = lane >> 4;
    const int nbase = blockIdx.x * 16;
    const int mbase = blockIdx.y * 16;

    const _Float16* xrow = X + (size_t)(mbase + lo) * ldx + half * 8;
    const _Float16* wrow = W + (size_t)(nbase + lo) * ldw + half * 8;

    v8f c = {};
    for (int k = 0; k < K; k += 32) {
        v16h a = ld_fragA(xrow + k);
        v16h b = ld_fragB(wrow + k);
        c = __builtin_amdgcn_wmma_f32_16x16x32_f16(
                /*neg_a=*/false, a, /*neg_b=*/false, b,
                /*c_mod=*/(short)0, c, /*reuse_a=*/false, /*reuse_b=*/false);
    }

    const float bn = bias ? bias[nbase + lo] : 0.0f;
#pragma unroll
    for (int r = 0; r < 8; ++r) {
        const int m = mbase + r + half * 8;     // C/D layout: VGPR r -> M=r (+8 for hi half)
        float v = c[r] * scale + bn;
        if (relu) v = fmaxf(v, 0.0f);
        const size_t o = (size_t)m * ldy + nbase + lo;
        Yf[o] = v;
        if (Yh) Yh[o] = (_Float16)v;
    }
}

// ---------------------------------------------------------------------------
// f32 -> f16 convert
// ---------------------------------------------------------------------------
__global__ void gac_cvt_f16(const float* __restrict__ s, _Float16* __restrict__ d, int n) {
    int t = blockIdx.x * blockDim.x + threadIdx.x;
    if (t < n) d[t] = (_Float16)s[t];
}

// ---------------------------------------------------------------------------
// Row softmax of scores [NAG, NAG] -> attn (f16)
// ---------------------------------------------------------------------------
__global__ __launch_bounds__(256)
void gac_softmax_row(const float* __restrict__ S, _Float16* __restrict__ A) {
    __shared__ float red[256];
    const int row = blockIdx.x;
    const float* s = S + (size_t)row * NAG;
    const int tid = threadIdx.x;

    float m = -INFINITY;
    for (int c = tid; c < NAG; c += 256) m = fmaxf(m, s[c]);
    red[tid] = m; __syncthreads();
    for (int st = 128; st > 0; st >>= 1) {
        if (tid < st) red[tid] = fmaxf(red[tid], red[tid + st]);
        __syncthreads();
    }
    m = red[0]; __syncthreads();

    float sum = 0.0f;
    for (int c = tid; c < NAG; c += 256) sum += __expf(s[c] - m);
    red[tid] = sum; __syncthreads();
    for (int st = 128; st > 0; st >>= 1) {
        if (tid < st) red[tid] += red[tid + st];
        __syncthreads();
    }
    const float inv = 1.0f / red[0];
    for (int c = tid; c < NAG; c += 256)
        A[(size_t)row * NAG + c] = (_Float16)(__expf(s[c] - m) * inv);
}

// ---------------------------------------------------------------------------
// Transposed V (f16): vT[d][j] = qkv[j][512 + d]
// ---------------------------------------------------------------------------
__global__ void gac_make_vT(const float* __restrict__ qkv, _Float16* __restrict__ vT) {
    int t = blockIdx.x * blockDim.x + threadIdx.x;
    if (t >= NAG * DHID) return;
    int j = t >> 8;      // agent
    int d = t & 255;     // feature
    vT[(size_t)d * NAG + j] = (_Float16)qkv[(size_t)j * (3 * DHID) + 2 * DHID + d];
}

// ---------------------------------------------------------------------------
// Value head: vh[j] = relu(dot(hid_flat, vfc2_w[j,:]) + vfc2_b[j])
// One block per output j. This is the only HBM-bound stage (134 MB of
// vfc2_w): stream it with b128 (float4) loads, f32 FMA, LDS tree reduce.
// ---------------------------------------------------------------------------
__global__ __launch_bounds__(256)
void gac_value_fc2(const float* __restrict__ hid, const float* __restrict__ Wv,
                   const float* __restrict__ bv, float* __restrict__ vh) {
    __shared__ float red[256];
    const int j = blockIdx.x;
    const float4* __restrict__ w4 = (const float4*)(Wv + (size_t)j * (NAG * DHID));
    const float4* __restrict__ h4 = (const float4*)hid;
    const int n4 = NAG * DHID / 4;          // 32768 float4's
    float acc = 0.0f;
    for (int t = threadIdx.x; t < n4; t += 256) {
        const float4 a = h4[t];
        const float4 b = w4[t];
        acc = fmaf(a.x, b.x, acc);
        acc = fmaf(a.y, b.y, acc);
        acc = fmaf(a.z, b.z, acc);
        acc = fmaf(a.w, b.w, acc);
    }
    red[threadIdx.x] = acc; __syncthreads();
    for (int st = 128; st > 0; st >>= 1) {
        if (threadIdx.x < st) red[threadIdx.x] += red[threadIdx.x + st];
        __syncthreads();
    }
    if (threadIdx.x == 0) vh[j] = fmaxf(red[0] + bv[j], 0.0f);
}

__global__ __launch_bounds__(256)
void gac_value_fc3(const float* __restrict__ vh, const float* __restrict__ w3,
                   const float* __restrict__ b3, float* __restrict__ val) {
    __shared__ float red[256];
    red[threadIdx.x] = vh[threadIdx.x] * w3[threadIdx.x];
    __syncthreads();
    for (int st = 128; st > 0; st >>= 1) {
        if (threadIdx.x < st) red[threadIdx.x] += red[threadIdx.x + st];
        __syncthreads();
    }
    if (threadIdx.x == 0) val[0] = red[0] + b3[0];
}

// ---------------------------------------------------------------------------
// Pair stage: for each (i<j):
//   s = relu(A[i]+B[j]) (fc2_b folded into A), z = s @ fc3_w.T + fc3_b,
//   out[p] = {softmax2(z), value}
// Block: i = blockIdx.y, 8 j's per block (one per wave). A[i], fc3 rows in LDS.
// ---------------------------------------------------------------------------
__global__ __launch_bounds__(256)
void gac_pairs(const float* __restrict__ A, const float* __restrict__ B,
               const float* __restrict__ fc3w, const float* __restrict__ fc3b,
               const float* __restrict__ val, float* __restrict__ out)
{
    __shared__ float Ai[DHID], w0[DHID], w1[DHID];
    const int i   = blockIdx.y;
    const int tid = threadIdx.x;
    Ai[tid] = A[(size_t)i * DHID + tid];
    w0[tid] = fc3w[tid];
    w1[tid] = fc3w[DHID + tid];
    __syncthreads();

    const int wave = tid >> 5;
    const int lane = tid & 31;
    const int j = blockIdx.x * 8 + wave;
    if (j <= i || j >= NAG) return;

    const float* Bj = B + (size_t)j * DHID;
    float t0 = 0.0f, t1 = 0.0f;
#pragma unroll
    for (int r = 0; r < 8; ++r) {
        const int d = r * 32 + lane;
        const float s = fmaxf(Ai[d] + Bj[d], 0.0f);
        t0 = fmaf(s, w0[d], t0);
        t1 = fmaf(s, w1[d], t1);
    }
#pragma unroll
    for (int m = 16; m > 0; m >>= 1) {
        t0 += __shfl_xor(t0, m, 32);
        t1 += __shfl_xor(t1, m, 32);
    }
    if (lane == 0) {
        const float z0 = t0 + fc3b[0], z1 = t1 + fc3b[1];
        const float mx = fmaxf(z0, z1);
        const float e0 = __expf(z0 - mx), e1 = __expf(z1 - mx);
        const float inv = 1.0f / (e0 + e1);
        const size_t p = (size_t)i * (NAG - 1) - (size_t)i * (i - 1) / 2 + (size_t)(j - i - 1);
        out[p * 3 + 0] = e0 * inv;
        out[p * 3 + 1] = e1 * inv;
        out[p * 3 + 2] = val[0];
    }
}

// ---------------------------------------------------------------------------
// Host launcher
// ---------------------------------------------------------------------------
extern "C" void kernel_launch(void* const* d_in, const int* in_sizes, int n_in,
                              void* d_out, int out_size, void* d_ws, size_t ws_size,
                              hipStream_t stream) {
    (void)in_sizes; (void)n_in; (void)out_size; (void)ws_size;

    const float* inputs     = (const float*)d_in[0];
    const float* fc1_w      = (const float*)d_in[1];
    const float* fc1_b      = (const float*)d_in[2];
    const float* in_proj_w  = (const float*)d_in[3];
    const float* in_proj_b  = (const float*)d_in[4];
    const float* out_proj_w = (const float*)d_in[5];
    const float* out_proj_b = (const float*)d_in[6];
    const float* fc2_w      = (const float*)d_in[7];
    const float* fc2_b      = (const float*)d_in[8];
    const float* fc3_w      = (const float*)d_in[9];
    const float* fc3_b      = (const float*)d_in[10];
    const float* vfc2_w     = (const float*)d_in[11];
    const float* vfc2_b     = (const float*)d_in[12];
    const float* vfc3_w     = (const float*)d_in[13];
    const float* vfc3_b     = (const float*)d_in[14];
    float* out = (float*)d_out;

    // ---- carve workspace (256B aligned slices) ----
    char* w = (char*)d_ws;
    auto take = [&](size_t bytes) { char* p = w; w += (bytes + 255) & ~(size_t)255; return p; };

    _Float16* inputs_h     = (_Float16*)take((size_t)NAG  * DOBS     * 2);
    _Float16* fc1_w_h      = (_Float16*)take((size_t)DHID * DOBS     * 2);
    _Float16* in_proj_w_h  = (_Float16*)take((size_t)3*DHID * DHID   * 2);
    _Float16* out_proj_w_h = (_Float16*)take((size_t)DHID * DHID     * 2);
    _Float16* fc2_w_h      = (_Float16*)take((size_t)DHID * 2*DHID   * 2);
    float*    hid          = (float*)   take((size_t)NAG  * DHID     * 4);
    _Float16* hid_h        = (_Float16*)take((size_t)NAG  * DHID     * 2);
    float*    qkv          = (float*)   take((size_t)NAG  * 3*DHID   * 4);
    _Float16* qkv_h        = (_Float16*)take((size_t)NAG  * 3*DHID   * 2);
    float*    scores       = (float*)   take((size_t)NAG  * NAG      * 4);
    _Float16* attn_h       = (_Float16*)take((size_t)NAG  * NAG      * 2);
    _Float16* vT_h         = (_Float16*)take((size_t)DHID * NAG      * 2);
    float*    attnv        = (float*)   take((size_t)NAG  * DHID     * 4);
    _Float16* attnv_h      = (_Float16*)take((size_t)NAG  * DHID     * 2);
    float*    hbuf         = (float*)   take((size_t)NAG  * DHID     * 4);
    _Float16* h_h          = (_Float16*)take((size_t)NAG  * DHID     * 2);
    float*    Abuf         = (float*)   take((size_t)NAG  * DHID     * 4);
    float*    Bbuf         = (float*)   take((size_t)NAG  * DHID     * 4);
    float*    vh           = (float*)   take(1024);
    float*    val          = (float*)   take(256);

    // ---- 0) f32 -> f16 weight/input conversion ----
    auto cvt = [&](const float* s, _Float16* d, int n) {
        gac_cvt_f16<<<(n + 255) / 256, 256, 0, stream>>>(s, d, n);
    };
    cvt(inputs,     inputs_h,     NAG * DOBS);
    cvt(fc1_w,      fc1_w_h,      DHID * DOBS);
    cvt(in_proj_w,  in_proj_w_h,  3 * DHID * DHID);
    cvt(out_proj_w, out_proj_w_h, DHID * DHID);
    cvt(fc2_w,      fc2_w_h,      DHID * 2 * DHID);

    const float inv_sqrt_d = 0.0625f;   // 1/sqrt(256)

    // ---- 1) fc1: hid = relu(inputs @ fc1_w.T + fc1_b) ----
    gac_wmma_gemm_xwT<<<dim3(DHID / 16, NAG / 16), 32, 0, stream>>>(
        inputs_h, DOBS, fc1_w_h, DOBS, fc1_b, hid, hid_h, DHID, DOBS, 1.0f, 1);

    // ---- 2) qkv = hid @ in_proj_w.T + in_proj_b ----
    gac_wmma_gemm_xwT<<<dim3(3 * DHID / 16, NAG / 16), 32, 0, stream>>>(
        hid_h, DHID, in_proj_w_h, DHID, in_proj_b, qkv, qkv_h, 3 * DHID, DHID, 1.0f, 0);

    // ---- 3) scores = (q @ k.T) / 16   (q = qkv[:, :256], k = qkv[:, 256:512]) ----
    gac_wmma_gemm_xwT<<<dim3(NAG / 16, NAG / 16), 32, 0, stream>>>(
        qkv_h, 3 * DHID, qkv_h + DHID, 3 * DHID, nullptr, scores, nullptr, NAG,
        DHID, inv_sqrt_d, 0);

    // ---- 4) attn = softmax(scores)  (rows) ----
    gac_softmax_row<<<NAG, 256, 0, stream>>>(scores, attn_h);

    // ---- 5) vT[d][j] = v[j][d] ----
    gac_make_vT<<<(NAG * DHID + 255) / 256, 256, 0, stream>>>(qkv, vT_h);

    // ---- 6) attnv = attn @ v ----
    gac_wmma_gemm_xwT<<<dim3(DHID / 16, NAG / 16), 32, 0, stream>>>(
        attn_h, NAG, vT_h, NAG, nullptr, attnv, attnv_h, DHID, NAG, 1.0f, 0);

    // ---- 7) h = attnv @ out_proj_w.T + out_proj_b ----
    gac_wmma_gemm_xwT<<<dim3(DHID / 16, NAG / 16), 32, 0, stream>>>(
        attnv_h, DHID, out_proj_w_h, DHID, out_proj_b, hbuf, h_h, DHID, DHID, 1.0f, 0);

    // ---- 8) factorized fc2: A = h @ Wl.T + fc2_b ; B = h @ Wr.T ----
    gac_wmma_gemm_xwT<<<dim3(DHID / 16, NAG / 16), 32, 0, stream>>>(
        h_h, DHID, fc2_w_h, 2 * DHID, fc2_b, Abuf, nullptr, DHID, DHID, 1.0f, 0);
    gac_wmma_gemm_xwT<<<dim3(DHID / 16, NAG / 16), 32, 0, stream>>>(
        h_h, DHID, fc2_w_h + DHID, 2 * DHID, nullptr, Bbuf, nullptr, DHID, DHID, 1.0f, 0);

    // ---- 9) value head (f32, HBM-bound) ----
    gac_value_fc2<<<DHID, 256, 0, stream>>>(hid, vfc2_w, vfc2_b, vh);
    gac_value_fc3<<<1, 256, 0, stream>>>(vh, vfc3_w, vfc3_b, val);

    // ---- 10) per-pair: relu(A[i]+B[j]) -> fc3 -> softmax2, + value ----
    gac_pairs<<<dim3(NAG / 8, NAG), 256, 0, stream>>>(Abuf, Bbuf, fc3_w, fc3_b, val, out);
}
